// DeepFM_88304527606398
// MI455X (gfx1250) — matrix operand order repaired and checked by reading, not compiled, not run on hardware
//
#include <hip/hip_runtime.h>
#include <hip/hip_bf16.h>
#include <cmath>

typedef __attribute__((ext_vector_type(16))) _Float16 v16h;
typedef __attribute__((ext_vector_type(8)))  _Float16 v8h;
typedef __attribute__((ext_vector_type(8)))  float    v8f;
typedef __attribute__((ext_vector_type(4)))  float    f4;

#define BATCH 4096
#define NUMF  13
#define NCAT  26
#define VOC   1000
#define EMB   16
#define D_IN  624      // 13*16 + 26*16
#define KPAD1 640      // D_IN padded to multiple of 64 (ping-pong K-step)
#define H1    512
#define H2    256

// ---------------------------------------------------------------------------
// Weight prep: fp32 W (K x N, row-major) -> f16 W^T (N x Kpad), zero K-pad.
// ---------------------------------------------------------------------------
__global__ void __launch_bounds__(256)
cvt_transpose(const float* __restrict__ W, _Float16* __restrict__ Wt,
              int K, int N, int Kpad) {
    int total = N * Kpad;
    for (int i = blockIdx.x * 256 + threadIdx.x; i < total; i += gridDim.x * 256) {
        int n = i / Kpad;
        int k = i - n * Kpad;
        float v = (k < K) ? W[(size_t)k * N + n] : 0.0f;
        Wt[(size_t)n * Kpad + k] = (_Float16)v;
    }
}

// ---------------------------------------------------------------------------
// Per-sample: one-hot scan (HBM-bound, 426 MB total, read-once -> NT loads so
// the stream does not thrash the 192 MB L2 that holds weights/activations),
// embedding assembly into f16 (row-major, Kpad=640, zero-padded), FM terms.
// One 256-thread block per batch row.
// ---------------------------------------------------------------------------
__global__ void __launch_bounds__(256)
embed_fm(const float* __restrict__ xnum, const float* __restrict__ oh,
         const float* __restrict__ W_num, const float* __restrict__ b_num,
         const float* __restrict__ emb_tables, const float* __restrict__ w_fm,
         const float* __restrict__ b_fm, _Float16* __restrict__ embed,
         float* __restrict__ y_fm) {
    __shared__ int   labels[NCAT];
    __shared__ float sums[2 * EMB];   // [0..15] = num field sum, [16..31] = cat field sum

    const int b   = blockIdx.x;
    const int tid = threadIdx.x;

    if (tid < NCAT)     labels[tid] = 0;
    if (tid < 2 * EMB)  sums[tid]   = 0.0f;
    __syncthreads();

    // --- vectorized non-temporal one-hot scan (global_load_b128, TH=NT)
    const f4* row4 = (const f4*)(oh + (size_t)b * (NCAT * VOC));
    for (int i = tid; i < (NCAT * VOC) / 4; i += 256) {
        f4 v = __builtin_nontemporal_load(&row4[i]);
        int base = i * 4;
        if (v.x > 0.5f) labels[(base + 0) / VOC] = (base + 0) % VOC;
        if (v.y > 0.5f) labels[(base + 1) / VOC] = (base + 1) % VOC;
        if (v.z > 0.5f) labels[(base + 2) / VOC] = (base + 2) % VOC;
        if (v.w > 0.5f) labels[(base + 3) / VOC] = (base + 3) % VOC;
    }
    __syncthreads();

    _Float16* erow = embed + (size_t)b * KPAD1;

    // --- numeric embedding: num_vec[f,e] = x[f]*W_num[f,e] + b_num[f,e]
    if (tid < NUMF * EMB) {
        int f = tid / EMB, e = tid % EMB;
        float v = xnum[(size_t)b * NUMF + f] * W_num[f * EMB + e] + b_num[f * EMB + e];
        erow[tid] = (_Float16)v;
        atomicAdd(&sums[e], v);              // ds_add_f32
    }
    // --- categorical embedding gather (tables ~1.7 MB, L2-resident)
    for (int idx = tid; idx < NCAT * EMB; idx += 256) {
        int c = idx / EMB, e = idx % EMB;
        float v = emb_tables[((size_t)c * VOC + labels[c]) * EMB + e];
        erow[NUMF * EMB + idx] = (_Float16)v;
        atomicAdd(&sums[EMB + e], v);
    }
    // --- zero pad K 624..639 so WMMA K-loop stays exact
    if (tid < KPAD1 - D_IN) erow[D_IN + tid] = (_Float16)0.0f;
    __syncthreads();

    if (tid == 0) {
        float fe = 0.0f;
        for (int e = 0; e < EMB; ++e) fe += sums[e] * sums[EMB + e];
        float sp = 0.0f;
        for (int f = 0; f < NUMF; ++f) sp += xnum[(size_t)b * NUMF + f] * w_fm[f];
        for (int c = 0; c < NCAT; ++c) sp += w_fm[NUMF + c * VOC + labels[c]];
        y_fm[b] = fe + sp + b_fm[0];
    }
}

// ---------------------------------------------------------------------------
// WMMA f16 GEMM + bias + ReLU, f16 output (M x N row-major, ldc = N).
//   C[M,N] = relu(A[M,Kpad] * Bt[N,Kpad]^T + bias[N]);  Kpad % 64 == 0.
// One wave computes a 16x64 strip (A-fragment reused across 4 WMMA tiles).
// Ping-pong software pipeline (K-step 64 = two WMMA groups): each fragment
// buffer is only ever written by loads, never by register copies, so the
// compiler renames instead of emitting v_mov chains; loads for one buffer
// overlap the other buffer's WMMAs.
// Fragment layouts per ISA 16-bit tables: A lane half selects K 0..7/8..15
// (V0-3) and K 16..23/24..31 (V4-7); B uses the same K order for one column.
// ---------------------------------------------------------------------------
__global__ void __launch_bounds__(256)
gemm_wmma_bias_relu(const _Float16* __restrict__ A, const _Float16* __restrict__ Bt,
                    const float* __restrict__ bias, _Float16* __restrict__ C,
                    int M, int N, int Kpad) {
    const int lane = threadIdx.x & 31;
    const int half = lane >> 4;
    const int l16  = lane & 15;
    const int wave = (blockIdx.x * 256 + threadIdx.x) >> 5;
    const int mtiles = M >> 4;
    const int tileM = wave % mtiles;
    const int tileN = wave / mtiles;          // 64-wide N strip index

    const _Float16* arow = A  + (size_t)(tileM * 16 + l16) * Kpad;
    const _Float16* bcol = Bt + (size_t)(tileN * 64 + l16) * Kpad;

    auto loadA = [&](int k0) -> v16h {
        v8h alo = *(const v8h*)(arow + k0 + half * 8);
        v8h ahi = *(const v8h*)(arow + k0 + 16 + half * 8);
        return __builtin_shufflevector(alo, ahi,
                   0,1,2,3,4,5,6,7,8,9,10,11,12,13,14,15);
    };
    auto loadB = [&](int k0, int nt) -> v16h {
        const _Float16* bp = bcol + (size_t)nt * 16 * Kpad + k0;
        v8h blo = *(const v8h*)(bp + half * 8);
        v8h bhi = *(const v8h*)(bp + 16 + half * 8);
        return __builtin_shufflevector(blo, bhi,
                   0,1,2,3,4,5,6,7,8,9,10,11,12,13,14,15);
    };

    v8f acc[4] = {v8f{}, v8f{}, v8f{}, v8f{}};

    // prologue: fill ping buffer (K-step 0)
    v16h aP = loadA(0);
    v16h bP[4];
#pragma unroll
    for (int nt = 0; nt < 4; ++nt) bP[nt] = loadB(0, nt);

    int k0 = 0;
    for (; k0 + 64 < Kpad; k0 += 64) {
        // fill pong buffer (k0+32) while ping WMMAs run
        v16h aQ = loadA(k0 + 32);
        v16h bQ[4];
#pragma unroll
        for (int nt = 0; nt < 4; ++nt) bQ[nt] = loadB(k0 + 32, nt);
#pragma unroll
        for (int nt = 0; nt < 4; ++nt)
            acc[nt] = __builtin_amdgcn_wmma_f32_16x16x32_f16(
                          false, aP, false, bP[nt], (short)0, acc[nt],
                          false, false);
        // refill ping buffer (k0+64) while pong WMMAs run
        aP = loadA(k0 + 64);
#pragma unroll
        for (int nt = 0; nt < 4; ++nt) bP[nt] = loadB(k0 + 64, nt);
#pragma unroll
        for (int nt = 0; nt < 4; ++nt)
            acc[nt] = __builtin_amdgcn_wmma_f32_16x16x32_f16(
                          false, aQ, false, bQ[nt], (short)0, acc[nt],
                          false, false);
    }
    // tail: last 64-wide K-step (k0 and k0+32)
    {
        v16h aQ = loadA(k0 + 32);
        v16h bQ[4];
#pragma unroll
        for (int nt = 0; nt < 4; ++nt) bQ[nt] = loadB(k0 + 32, nt);
#pragma unroll
        for (int nt = 0; nt < 4; ++nt)
            acc[nt] = __builtin_amdgcn_wmma_f32_16x16x32_f16(
                          false, aP, false, bP[nt], (short)0, acc[nt],
                          false, false);
#pragma unroll
        for (int nt = 0; nt < 4; ++nt)
            acc[nt] = __builtin_amdgcn_wmma_f32_16x16x32_f16(
                          false, aQ, false, bQ[nt], (short)0, acc[nt],
                          false, false);
    }

    // Epilogue: C/D layout -> VGPR i holds row (tileM*16 + i + half*8), col = l16
    const int rowBase = tileM * 16 + half * 8;
#pragma unroll
    for (int nt = 0; nt < 4; ++nt) {
        const int col = tileN * 64 + nt * 16 + l16;
        const float bv = bias[col];
#pragma unroll
        for (int i = 0; i < 8; ++i) {
            float v = acc[nt][i] + bv;
            v = v > 0.0f ? v : 0.0f;
            C[(size_t)(rowBase + i) * N + col] = (_Float16)v;
        }
    }
}

// ---------------------------------------------------------------------------
// Final: y_deep = relu(h2 . W3 + b3); out = sigmoid(y_fm + y_deep).
// One wave per row, shuffle reduction.
// ---------------------------------------------------------------------------
__global__ void __launch_bounds__(256)
final_matvec(const _Float16* __restrict__ h2, const float* __restrict__ W3,
             const float* __restrict__ b3, const float* __restrict__ y_fm,
             float* __restrict__ out) {
    const int row  = blockIdx.x * 8 + (threadIdx.x >> 5);
    const int lane = threadIdx.x & 31;
    float p = 0.0f;
    for (int j = lane; j < H2; j += 32)
        p += (float)h2[(size_t)row * H2 + j] * W3[j];
#pragma unroll
    for (int o = 16; o > 0; o >>= 1)
        p += __shfl_xor(p, o, 32);
    if (lane == 0) {
        float yd = p + b3[0];
        yd = yd > 0.0f ? yd : 0.0f;
        float s = y_fm[row] + yd;
        out[row] = 1.0f / (1.0f + expf(-s));
    }
}

// ---------------------------------------------------------------------------
extern "C" void kernel_launch(void* const* d_in, const int* in_sizes, int n_in,
                              void* d_out, int out_size, void* d_ws, size_t ws_size,
                              hipStream_t stream) {
    const float* xnum   = (const float*)d_in[0];   // (4096,13)
    const float* catoh  = (const float*)d_in[1];   // (4096,26000)
    const float* W_num  = (const float*)d_in[2];   // (13,16)
    const float* b_num  = (const float*)d_in[3];   // (13,16)
    const float* embtab = (const float*)d_in[4];   // (26,1000,16)
    const float* w_fm   = (const float*)d_in[5];   // (26013,1)
    const float* b_fm   = (const float*)d_in[6];   // (1,)
    const float* W1     = (const float*)d_in[7];   // (624,512)
    const float* b1     = (const float*)d_in[8];   // (512,)
    const float* W2     = (const float*)d_in[9];   // (512,256)
    const float* b2     = (const float*)d_in[10];  // (256,)
    const float* W3     = (const float*)d_in[11];  // (256,1)
    const float* b3     = (const float*)d_in[12];  // (1,)
    float* out = (float*)d_out;                    // (4096,1) fp32

    // Workspace layout (all offsets 256B-aligned)
    char* ws = (char*)d_ws;
    size_t off = 0;
    _Float16* embed = (_Float16*)(ws + off); off += (size_t)BATCH * KPAD1 * 2;  // 5 MB
    _Float16* W1T   = (_Float16*)(ws + off); off += (size_t)H1 * KPAD1 * 2;     // 640 KB
    _Float16* W2T   = (_Float16*)(ws + off); off += (size_t)H2 * H1 * 2;        // 256 KB
    _Float16* h1    = (_Float16*)(ws + off); off += (size_t)BATCH * H1 * 2;     // 4 MB
    _Float16* h2    = (_Float16*)(ws + off); off += (size_t)BATCH * H2 * 2;     // 2 MB
    float*    yfm   = (float*)(ws + off);    off += (size_t)BATCH * 4;          // 16 KB

    // 1) weight conversion (tiny)
    cvt_transpose<<<(H1 * KPAD1 + 255) / 256, 256, 0, stream>>>(W1, W1T, D_IN, H1, KPAD1);
    cvt_transpose<<<(H2 * H1   + 255) / 256, 256, 0, stream>>>(W2, W2T, H1,   H2, H1);

    // 2) one-hot scan + embedding + FM (HBM-bound: 426 MB streamed, NT)
    embed_fm<<<BATCH, 256, 0, stream>>>(xnum, catoh, W_num, b_num, embtab,
                                        w_fm, b_fm, embed, yfm);

    // 3) MLP layer 1: (4096 x 640) * (640 x 512) -> waves = 256 * 8
    gemm_wmma_bias_relu<<<(BATCH / 16) * (H1 / 64) / 8, 256, 0, stream>>>(
        embed, W1T, b1, h1, BATCH, H1, KPAD1);

    // 4) MLP layer 2: (4096 x 512) * (512 x 256) -> waves = 256 * 4
    gemm_wmma_bias_relu<<<(BATCH / 16) * (H2 / 64) / 8, 256, 0, stream>>>(
        h1, W2T, b2, h2, BATCH, H2, H1);

    // 5) final matvec + sigmoid
    final_matvec<<<BATCH / 8, 256, 0, stream>>>(h2, W3, b3, yfm, out);

    (void)in_sizes; (void)n_in; (void)out_size; (void)ws_size;
}
